// RAN_50577534878247
// MI455X (gfx1250) — matrix operand order repaired and checked by reading, not compiled, not run on hardware
//
#include <hip/hip_runtime.h>

#define TT 128
#define BB 32
#define II 128
#define HH 128
#define VV 32000
#define NN (TT * BB) // 4096

typedef __bf16 bf16;
typedef __attribute__((ext_vector_type(16))) __bf16 v16bf;
typedef __attribute__((ext_vector_type(8)))  __bf16 v8bf;
typedef __attribute__((ext_vector_type(8)))  float  v8f;
typedef __attribute__((ext_vector_type(4)))  float  v4f;

static __device__ inline bf16 f2bf(float f) {
  unsigned u = __builtin_bit_cast(unsigned, f);
  u += 0x7FFFu + ((u >> 16) & 1u); // round-to-nearest-even
  unsigned short h = (unsigned short)(u >> 16);
  return __builtin_bit_cast(bf16, h);
}

// ---- WMMA fragment loaders (wave32, 16x16x32 bf16, contiguous k = 16*(lane/16) + s in A and B) ----
static __device__ inline v16bf load_a_frag(const bf16* base, int ld, int m0, int k0, int lane) {
  int row = m0 + (lane & 15);
  int kh  = (lane >> 4);
  return *(const v16bf*)(base + (size_t)row * ld + k0 + kh * 16);
}

static __device__ inline v16bf load_b_frag(const bf16* base, int ld, int n0, int k0, int lane) {
  int n  = n0 + (lane & 15);
  int kh = (lane >> 4);
  return *(const v16bf*)(base + (size_t)n * ld + k0 + kh * 16);
}

// ---- f32 -> bf16 conversion ----
__global__ void __launch_bounds__(256)
cvt_f32_bf16(const float* __restrict__ src, bf16* __restrict__ dst, int n) {
  int i = blockIdx.x * blockDim.x + threadIdx.x;
  int stride = gridDim.x * blockDim.x;
  for (; i < n; i += stride) dst[i] = f2bf(src[i]);
}

// ---- input projections: xc/xi/xf = X @ W^T + b   (X: [NN][II] bf16) ----
__global__ void __launch_bounds__(256)
proj_kernel(const bf16* __restrict__ X,
            const bf16* __restrict__ Wc, const bf16* __restrict__ Wi, const bf16* __restrict__ Wf,
            const float* __restrict__ bc, const float* __restrict__ bi, const float* __restrict__ bfv,
            float* __restrict__ xc, float* __restrict__ xi, float* __restrict__ xf) {
  int lane = threadIdx.x & 31;
  int wv   = threadIdx.x >> 5;
  int m0 = blockIdx.x * 16;
  int n0 = wv * 16;

  v16bf a[4];
#pragma unroll
  for (int kk = 0; kk < 4; ++kk) a[kk] = load_a_frag(X, II, m0, kk * 32, lane);

  const bf16*  Ws[3]   = {Wc, Wi, Wf};
  const float* bs[3]   = {bc, bi, bfv};
  float*       outs[3] = {xc, xi, xf};

  int col   = n0 + (lane & 15);
  int rbase = m0 + ((lane >> 4) << 3);
#pragma unroll
  for (int g = 0; g < 3; ++g) {
    v8f c = {};
#pragma unroll
    for (int kk = 0; kk < 4; ++kk) {
      v16bf b = load_b_frag(Ws[g], II, n0, kk * 32, lane);
      c = __builtin_amdgcn_wmma_f32_16x16x32_bf16(false, a[kk], false, b, (short)0, c, false, false);
    }
    float bias = bs[g][col];
    float* o = outs[g];
#pragma unroll
    for (int r = 0; r < 8; ++r) o[(size_t)(rbase + r) * HH + col] = c[r] + bias;
  }
}

// ---- recurrent scan: single workgroup, h lives in LDS, gate GEMMs on WMMA ----
__global__ void __launch_bounds__(256)
scan_kernel(const float* __restrict__ h0,
            const bf16* __restrict__ Wic, const bf16* __restrict__ Wfc,
            const float* __restrict__ b_ic, const float* __restrict__ b_fc,
            const float* __restrict__ xc, const float* __restrict__ xi, const float* __restrict__ xf,
            bf16* __restrict__ Hs, float* __restrict__ h_last) {
  __shared__ __attribute__((aligned(32))) bf16  sh[BB * HH];
  __shared__ __attribute__((aligned(32))) float shf[BB * HH];
  __shared__ __attribute__((aligned(32))) float sPi[BB * HH];
  __shared__ __attribute__((aligned(32))) float sPf[BB * HH];

  int tid  = threadIdx.x;
  int lane = tid & 31;
  int wv   = tid >> 5;

  for (int e = tid; e < BB * HH; e += 256) { float v = h0[e]; shf[e] = v; sh[e] = f2bf(v); }
  __syncthreads();

  for (int t = 0; t < TT; ++t) {
#pragma unroll
    for (int ti = 0; ti < 4; ++ti) {
      int tile = wv * 4 + ti;       // 32 tiles = gate(2) x mt(2) x nt(8)
      int gate = tile >> 4;
      int mt   = (tile >> 3) & 1;
      int nt   = tile & 7;
      int m0 = mt * 16, n0 = nt * 16;
      const bf16* W = gate ? Wfc : Wic;
      float*      P = gate ? sPf : sPi;
      v8f c = {};
#pragma unroll
      for (int kk = 0; kk < 4; ++kk) {
        v16bf a = load_a_frag(sh, HH, m0, kk * 32, lane);
        v16bf b = load_b_frag(W,  HH, n0, kk * 32, lane);
        c = __builtin_amdgcn_wmma_f32_16x16x32_bf16(false, a, false, b, (short)0, c, false, false);
      }
      int col   = n0 + (lane & 15);
      int rbase = m0 + ((lane >> 4) << 3);
#pragma unroll
      for (int r = 0; r < 8; ++r) P[(rbase + r) * HH + col] = c[r];
    }
    __syncthreads();

    const float* xct = xc + (size_t)t * BB * HH;
    const float* xit = xi + (size_t)t * BB * HH;
    const float* xft = xf + (size_t)t * BB * HH;
    bf16*        Ht  = Hs + (size_t)t * BB * HH;
    for (int e = tid; e < BB * HH; e += 256) {
      int hcol = e & (HH - 1);
      float iv = 1.f / (1.f + __expf(-(sPi[e] + b_ic[hcol] + xit[e])));
      float fv = 1.f / (1.f + __expf(-(sPf[e] + b_fc[hcol] + xft[e])));
      float cv = iv * xct[e] + fv * shf[e];
      float hv = tanhf(cv);
      shf[e] = hv;
      bf16 hb = f2bf(hv);
      sh[e] = hb;
      Ht[e] = hb;
    }
    __syncthreads();
  }
  for (int e = tid; e < BB * HH; e += 256) h_last[e] = shf[e];
}

// ---- logits = Hs @ lin_w^T + lin_b : 32x256 tile per block, A async-staged ----
#define MBLK 32
#define NBLK 256

__global__ void __launch_bounds__(256)
logits_kernel(const bf16* __restrict__ Hsq, const bf16* __restrict__ Wl,
              const float* __restrict__ bl, float* __restrict__ out) {
  __shared__ __attribute__((aligned(128))) bf16 sA[MBLK * HH]; // 8 KB
  int tid  = threadIdx.x;
  int lane = tid & 31;
  int wv   = tid >> 5;
  int mbase = blockIdx.y * MBLK;
  int nbase = blockIdx.x * NBLK;

  // Async-copy the contiguous 8 KB A tile (rows mbase..mbase+31) into LDS.
  // 256 lanes x 16 B x 2 iterations; tracked by ASYNCcnt.
  {
    const char* gsrc = (const char*)(Hsq + (size_t)mbase * HH);
    unsigned lds0 = (unsigned)(size_t)(&sA[0]);
#pragma unroll
    for (int it = 0; it < 2; ++it) {
      unsigned off = (unsigned)(tid * 16 + it * 4096);
      const char* g = gsrc + off;
      unsigned l = lds0 + off;
      asm volatile("global_load_async_to_lds_b128 %0, %1, off"
                   :: "v"(l), "v"(g) : "memory");
    }
    asm volatile("s_wait_asynccnt 0x0" ::: "memory");
  }
  __syncthreads();

  int mt = (wv >> 2) * 16;  // wave's M-tile within block
  int ng = (wv & 3);        // wave's group of 4 N-tiles
  v16bf a[4];
#pragma unroll
  for (int kk = 0; kk < 4; ++kk) a[kk] = load_a_frag(sA, HH, mt, kk * 32, lane);

  int rbase = mbase + mt + ((lane >> 4) << 3);
#pragma unroll
  for (int j = 0; j < 4; ++j) {
    int n0 = nbase + ng * 64 + j * 16;
    v8f c = {};
#pragma unroll
    for (int kk = 0; kk < 4; ++kk) {
      v16bf b = load_b_frag(Wl, HH, n0, kk * 32, lane);
      c = __builtin_amdgcn_wmma_f32_16x16x32_bf16(false, a[kk], false, b, (short)0, c, false, false);
    }
    int col = n0 + (lane & 15);
    float bias = bl[col];
#pragma unroll
    for (int r = 0; r < 8; ++r)
      __builtin_nontemporal_store(c[r] + bias,
                                  &out[(size_t)(rbase + r) * VV + col]);
  }
}

// ---- fused log_softmax: whole row (128 KB) lives in LDS ----
__global__ void __launch_bounds__(256)
softmax_kernel(float* __restrict__ logits) {
  extern __shared__ __attribute__((aligned(16))) float srow[]; // VV floats
  __shared__ float red[256];
  int row = blockIdx.x;
  int tid = threadIdx.x;
  float* p  = logits + (size_t)row * VV;
  v4f*   p4 = (v4f*)p;
  v4f*   s4 = (v4f*)srow;

  float m = -3.402823466e38f;
  for (int i = tid; i < VV / 4; i += 256) {
    v4f v = __builtin_nontemporal_load(&p4[i]);
    s4[i] = v;
    m = fmaxf(m, fmaxf(fmaxf(v.x, v.y), fmaxf(v.z, v.w)));
  }
  red[tid] = m; __syncthreads();
  for (int s = 128; s > 0; s >>= 1) { if (tid < s) red[tid] = fmaxf(red[tid], red[tid + s]); __syncthreads(); }
  float mx = red[0]; __syncthreads();

  float sum = 0.f;
  for (int i = tid; i < VV / 4; i += 256) {
    v4f v = s4[i];
    sum += __expf(v.x - mx) + __expf(v.y - mx) + __expf(v.z - mx) + __expf(v.w - mx);
  }
  red[tid] = sum; __syncthreads();
  for (int s = 128; s > 0; s >>= 1) { if (tid < s) red[tid] += red[tid + s]; __syncthreads(); }
  float lse = mx + __logf(red[0]);
  __syncthreads();

  for (int i = tid; i < VV / 4; i += 256) {
    v4f v = s4[i];
    v4f o; o.x = v.x - lse; o.y = v.y - lse; o.z = v.z - lse; o.w = v.w - lse;
    __builtin_nontemporal_store(o, &p4[i]);
  }
}

extern "C" void kernel_launch(void* const* d_in, const int* in_sizes, int n_in,
                              void* d_out, int out_size, void* d_ws, size_t ws_size,
                              hipStream_t stream) {
  (void)in_sizes; (void)n_in; (void)out_size; (void)ws_size;
  const float* input = (const float*)d_in[0];
  const float* h0    = (const float*)d_in[1];
  const float* w_cx  = (const float*)d_in[2];
  const float* w_ic  = (const float*)d_in[3];
  const float* w_ix  = (const float*)d_in[4];
  const float* w_fc  = (const float*)d_in[5];
  const float* w_fx  = (const float*)d_in[6];
  const float* b_cx  = (const float*)d_in[7];
  const float* b_ic  = (const float*)d_in[8];
  const float* b_ix  = (const float*)d_in[9];
  const float* b_fc  = (const float*)d_in[10];
  const float* b_fx  = (const float*)d_in[11];
  const float* lin_w = (const float*)d_in[12];
  const float* lin_b = (const float*)d_in[13];
  (void)b_cx; (void)b_fx; // folded: b_cx into xc via proj (passed below), b_fx likewise

  float* logp   = (float*)d_out;              // [NN][VV]
  float* h_last = logp + (size_t)NN * VV;     // [1,B,H] f32

  char* wp = (char*)d_ws;
  auto alloc = [&](size_t bytes) { char* p = wp; wp += (bytes + 255) & ~(size_t)255; return p; };
  bf16*  Xbf   = (bf16*)alloc((size_t)NN * II * 2);
  bf16*  Wcxb  = (bf16*)alloc((size_t)HH * II * 2);
  bf16*  Wixb  = (bf16*)alloc((size_t)HH * II * 2);
  bf16*  Wfxb  = (bf16*)alloc((size_t)HH * II * 2);
  bf16*  Wicb  = (bf16*)alloc((size_t)HH * HH * 2);
  bf16*  Wfcb  = (bf16*)alloc((size_t)HH * HH * 2);
  bf16*  Wlb   = (bf16*)alloc((size_t)VV * HH * 2);
  float* xc    = (float*)alloc((size_t)NN * HH * 4);
  float* xi    = (float*)alloc((size_t)NN * HH * 4);
  float* xf    = (float*)alloc((size_t)NN * HH * 4);
  bf16*  Hs    = (bf16*)alloc((size_t)NN * HH * 2);

  auto cvt = [&](const float* s, bf16* d, long long n) {
    int blocks = (int)((n + 255) / 256); if (blocks > 4096) blocks = 4096;
    cvt_f32_bf16<<<dim3(blocks), dim3(256), 0, stream>>>(s, d, (int)n);
  };
  cvt(input, Xbf, (long long)NN * II);
  cvt(w_cx, Wcxb, (long long)HH * II);
  cvt(w_ix, Wixb, (long long)HH * II);
  cvt(w_fx, Wfxb, (long long)HH * II);
  cvt(w_ic, Wicb, (long long)HH * HH);
  cvt(w_fc, Wfcb, (long long)HH * HH);
  cvt(lin_w, Wlb, (long long)VV * HH);

  proj_kernel<<<dim3(NN / 16), dim3(256), 0, stream>>>(
      Xbf, Wcxb, Wixb, Wfxb, d_in ? (const float*)d_in[7] : b_cx,
      (const float*)d_in[9], (const float*)d_in[11], xc, xi, xf);

  scan_kernel<<<dim3(1), dim3(256), 0, stream>>>(
      h0, Wicb, Wfcb, b_ic, b_fc, xc, xi, xf, Hs, h_last);

  logits_kernel<<<dim3(VV / NBLK, NN / MBLK), dim3(256), 0, stream>>>(Hs, Wlb, lin_b, logp);

  softmax_kernel<<<dim3(NN), dim3(256), (size_t)VV * 4, stream>>>(logp);
}
